// MinkUNet_78821239816592
// MI455X (gfx1250) — matrix operand order repaired and checked by reading, not compiled, not run on hardware
//
#include <hip/hip_runtime.h>
#include <hip/hip_bf16.h>

// ---------------------------------------------------------------------------
// MinkUNet-style dense 3D U-Net forward on MI455X (gfx1250).
// Implicit-GEMM convs on v_wmma_f32_16x16x32_bf16 (wave32, fp32 accumulate),
// 2x2 WMMA tiles per wave, LDS double buffering, im2col gather with hoisted voxel decode.
// Activations channels-last; weights packed to bf16 [K x Npad] panels with
// k = ((dz*kh+dy)*kw+dx)*cin + ci.
// ---------------------------------------------------------------------------

typedef __bf16 bf16_t;
typedef __attribute__((ext_vector_type(16))) __bf16 v16bf;
typedef __attribute__((ext_vector_type(8)))  float  v8f;

struct ConvDesc {
  int Din, Hin, Win;      // input spatial dims
  int Dout, Hout, Wout;   // output spatial dims (up mode: 2*in)
  int cin, cout;
  int kd, kh, kw;
  int stride, pad;
  int up, udz, udy, udx;  // up=1: tconv parity-offset scatter mode (1x1 GEMM)
  int Ktot, Npad;         // Npad multiple of block-N
  int M;                  // GEMM rows (out voxels; up mode: in voxels)
};

// ---------------------------------------------------------------------------
// Implicit-GEMM conv. Block tile BM x BN (64x128 or 128x64), 8 waves, each
// wave owns a 32x32 region = 2x2 WMMA tiles. K stepped by 32 through
// double-buffered LDS tiles. As is [m][k] row-major, Bs is [n][k] so both
// fragment loads are contiguous b128 pairs.
// ---------------------------------------------------------------------------
template <int BM, int BN>
__global__ __launch_bounds__(256) void conv_wmma(
    const bf16_t* __restrict__ A, const bf16_t* __restrict__ Bw,
    const float* __restrict__ bias, float* __restrict__ out, ConvDesc d)
{
  __shared__ bf16_t As[2][BM][40];
  __shared__ bf16_t Bs[2][BN][40];

  const int tid  = threadIdx.x;
  const int m0   = blockIdx.x * BM;
  const int n0   = blockIdx.y * BN;
  const int lane = tid & 31;
  const int wv   = tid >> 5;
  const int half = lane >> 4;
  const int l    = lane & 15;
  const int mw   = (BN == 128) ? (wv & 1) : (wv & 3);
  const int nw   = (BN == 128) ? (wv >> 1) : (wv >> 2);
  const int khw  = d.kh * d.kw;
  const int kq   = tid & 3;            // 8-wide K chunk owned for A staging
  constexpr int ROWS = BM / 64;        // A rows staged per thread

  // Hoist per-row voxel decode out of the K loop.
  int ozr[ROWS], oyr[ROWS], oxr[ROWS];
  #pragma unroll
  for (int r = 0; r < ROWS; ++r) {
    int gm = m0 + (tid >> 2) + r * 64;
    int HW = d.Hout * d.Wout;
    int oz = gm / HW;
    int rm = gm - oz * HW;
    int oy = rm / d.Wout;
    ozr[r] = oz; oyr[r] = oy; oxr[r] = rm - oy * d.Wout;
  }

  // Base index of the gather for window offset `off` (-1 if out of bounds).
  auto nbase = [&](int off, int oz, int oy, int ox) -> int {
    int dz = off / khw; int r2 = off - dz * khw;
    int dy = r2 / d.kw, dx = r2 - dy * d.kw;
    int iz = oz * d.stride - d.pad + dz;
    int iy = oy * d.stride - d.pad + dy;
    int ix = ox * d.stride - d.pad + dx;
    if ((unsigned)iz < (unsigned)d.Din && (unsigned)iy < (unsigned)d.Hin &&
        (unsigned)ix < (unsigned)d.Win)
      return ((iz * d.Hin + iy) * d.Win + ix) * d.cin;
    return -1;
  };

  auto stage = [&](int ks, int buf) {
    const int kk0 = ks << 5;
    // ---- B tile: 32 K-rows x BN cols, stored transposed Bs[n][k] ----
    #pragma unroll
    for (int rep = 0; rep < BN / 64; ++rep) {
      int k  = tid & 31;
      int nc = (tid >> 5) + rep * 8;          // 8-wide N chunk
      int gk = kk0 + k;
      uint4 v = {0u, 0u, 0u, 0u};
      if (gk < d.Ktot)
        v = *(const uint4*)(Bw + (size_t)gk * d.Npad + n0 + nc * 8);
      const bf16_t* pv = (const bf16_t*)&v;
      #pragma unroll
      for (int j = 0; j < 8; ++j) Bs[buf][nc * 8 + j][k] = pv[j];
    }
    // ---- A tile: BM rows x 32 K, gathered from the conv window ----
    #pragma unroll
    for (int r = 0; r < ROWS; ++r) {
      int mi = (tid >> 2) + r * 64;
      int gm = m0 + mi;
      alignas(16) bf16_t tmp[8];
      int kk = kk0 + kq * 8;
      if (gm < d.M) {
        if (d.up) {
          #pragma unroll
          for (int j = 0; j < 8; ++j)
            tmp[j] = (kk + j < d.Ktot) ? A[(size_t)gm * d.cin + kk + j]
                                       : (bf16_t)0.f;
        } else {
          int off  = kk / d.cin;
          int ci   = kk - off * d.cin;
          int base = nbase(off, ozr[r], oyr[r], oxr[r]);
          #pragma unroll
          for (int j = 0; j < 8; ++j) {
            tmp[j] = (kk + j < d.Ktot && base >= 0) ? A[(size_t)base + ci]
                                                    : (bf16_t)0.f;
            if (++ci == d.cin) { ci = 0; ++off; base = nbase(off, ozr[r], oyr[r], oxr[r]); }
          }
        }
      } else {
        #pragma unroll
        for (int j = 0; j < 8; ++j) tmp[j] = (bf16_t)0.f;
      }
      *(uint4*)&As[buf][mi][kq * 8] = *(const uint4*)tmp;
    }
  };

  v8f acc[2][2] = {{{}, {}}, {{}, {}}};

  auto compute = [&](int buf) {
    v16bf af[2], bfr[2];
    #pragma unroll
    for (int s = 0; s < 2; ++s) {
      int row = mw * 32 + s * 16 + l;
      uint4* pa = (uint4*)&af[s];
      pa[0] = *(const uint4*)&As[buf][row][half * 8];
      pa[1] = *(const uint4*)&As[buf][row][16 + half * 8];
      int col = nw * 32 + s * 16 + l;
      uint4* pb = (uint4*)&bfr[s];
      pb[0] = *(const uint4*)&Bs[buf][col][half * 8];
      pb[1] = *(const uint4*)&Bs[buf][col][16 + half * 8];
    }
    #pragma unroll
    for (int si = 0; si < 2; ++si)
      #pragma unroll
      for (int sj = 0; sj < 2; ++sj)
        acc[si][sj] = __builtin_amdgcn_wmma_f32_16x16x32_bf16(
            false, af[si], false, bfr[sj], (short)0, acc[si][sj], false, false);
  };

  const int ksteps = (d.Ktot + 31) >> 5;
  stage(0, 0);
  __syncthreads();
  for (int ks = 0; ks < ksteps; ++ks) {
    int cur = ks & 1;
    if (ks + 1 < ksteps) {
      __builtin_prefetch(Bw + (size_t)((ks + 2) << 5) * d.Npad + n0, 0, 1);
      stage(ks + 1, cur ^ 1);
    }
    compute(cur);
    __syncthreads();
  }

  // ---- store D tiles (lane0-15: M=r, lane16-31: M=8+r within a tile) ----
  #pragma unroll
  for (int si = 0; si < 2; ++si) {
    #pragma unroll
    for (int sj = 0; sj < 2; ++sj) {
      #pragma unroll
      for (int rr = 0; rr < 8; ++rr) {
        int ml = mw * 32 + si * 16 + half * 8 + rr;
        int nl = nw * 32 + sj * 16 + l;
        int gm = m0 + ml, gn = n0 + nl;
        if (gm < d.M && gn < d.cout) {
          float y = acc[si][sj][rr];
          if (bias) y += bias[gn];
          int ov;
          if (d.up) {
            int z = gm / (d.Hin * d.Win);
            int r = gm - z * d.Hin * d.Win;
            int yy = r / d.Win, xx = r - yy * d.Win;
            ov = ((2 * z + d.udz) * d.Hout + (2 * yy + d.udy)) * d.Wout +
                 (2 * xx + d.udx);
          } else {
            ov = gm;
          }
          out[(size_t)ov * d.cout + gn] = y;
        }
      }
    }
  }
}

// ---------------------------------------------------------------------------
// Support kernels
// ---------------------------------------------------------------------------
__global__ void to_cl_bf16(const float* __restrict__ x, int vox, int C,
                           bf16_t* __restrict__ dst) {
  int i = blockIdx.x * blockDim.x + threadIdx.x;
  if (i >= vox * C) return;
  int v = i / C, c = i - v * C;
  dst[i] = (bf16_t)x[(size_t)c * vox + v];   // NCDHW -> [vox][C]
}

// w (cout,cin,kd,kh,kw) -> bf16 [Kpad x Npad], k = off*cin + ci
__global__ void pack_w3d(const float* __restrict__ w, int cout, int cin, int kvol,
                         bf16_t* __restrict__ B, int Kpad, int Npad) {
  int i = blockIdx.x * blockDim.x + threadIdx.x;
  if (i >= Kpad * Npad) return;
  int n = i % Npad, k = i / Npad;
  float v = 0.f;
  int Kt = cin * kvol;
  if (k < Kt && n < cout) {
    int off = k / cin, ci = k - off * cin;
    v = w[(size_t)(n * cin + ci) * kvol + off];
  }
  B[i] = (bf16_t)v;
}

// tconv w (cin,cout,2,2,2) -> 8 slices of bf16 [Kpad x Npad], k = ci
__global__ void pack_wt(const float* __restrict__ w, int cin, int cout,
                        bf16_t* __restrict__ B, int Kpad, int Npad) {
  int i = blockIdx.x * blockDim.x + threadIdx.x;
  if (i >= 8 * Kpad * Npad) return;
  int n = i % Npad;
  int k = (i / Npad) % Kpad;
  int dd = i / (Npad * Kpad);
  float v = 0.f;
  if (k < cin && n < cout) v = w[(size_t)(k * cout + n) * 8 + dd];
  B[i] = (bf16_t)v;
}

// row-major (K,N) fp32 -> bf16 [Kpad x Npad]
__global__ void pack_rm(const float* __restrict__ w, int K, int N,
                        bf16_t* __restrict__ B, int Kpad, int Npad) {
  int i = blockIdx.x * blockDim.x + threadIdx.x;
  if (i >= Kpad * Npad) return;
  int n = i % Npad, k = i / Npad;
  B[i] = (bf16_t)((k < K && n < N) ? w[(size_t)k * N + n] : 0.f);
}

__global__ void bn_stats(const float* __restrict__ x, int vox, int C,
                         float* __restrict__ mean, float* __restrict__ var) {
  int c = blockIdx.x;
  float s = 0.f, s2 = 0.f;
  for (int v = threadIdx.x; v < vox; v += blockDim.x) {
    float t = x[(size_t)v * C + c];
    s += t; s2 += t * t;
  }
  __shared__ float sh[256], sh2[256];
  sh[threadIdx.x] = s; sh2[threadIdx.x] = s2;
  __syncthreads();
  for (int o = 128; o > 0; o >>= 1) {
    if (threadIdx.x < o) { sh[threadIdx.x] += sh[threadIdx.x + o]; sh2[threadIdx.x] += sh2[threadIdx.x + o]; }
    __syncthreads();
  }
  if (threadIdx.x == 0) {
    float m = sh[0] / vox;
    mean[c] = m;
    var[c]  = sh2[0] / vox - m * m;
  }
}

__global__ void bn_apply(const float* __restrict__ x, const float* __restrict__ mean,
                         const float* __restrict__ var, const float* __restrict__ g,
                         const float* __restrict__ b, const float* __restrict__ add,
                         int relu, float* __restrict__ of, bf16_t* __restrict__ ob,
                         int ob_stride, int ob_off, int vox, int C) {
  int i = blockIdx.x * blockDim.x + threadIdx.x;
  if (i >= vox * C) return;
  int v = i / C, c = i - v * C;
  float y = (x[i] - mean[c]) * rsqrtf(var[c] + 1e-5f) * g[c] + b[c];
  if (add) y += add[i];
  if (relu) y = fmaxf(y, 0.f);
  if (of) of[i] = y;
  if (ob) ob[(size_t)v * ob_stride + ob_off + c] = (bf16_t)y;
}

__global__ void copy_ch(const bf16_t* __restrict__ src, int Cs,
                        bf16_t* __restrict__ dst, int Cd, int off, int vox) {
  int i = blockIdx.x * blockDim.x + threadIdx.x;
  if (i >= vox * Cs) return;
  int v = i / Cs, c = i - v * Cs;
  dst[(size_t)v * Cd + off + c] = src[i];
}

// ---------------------------------------------------------------------------
// Host-side orchestration
// ---------------------------------------------------------------------------
struct Ctx {
  hipStream_t s;
  float *t0, *t2, *mean, *var;
  bf16_t *hbf, *Wp;
};

static inline int rnd(int v, int a) { return (v + a - 1) / a * a; }
static inline int blkN(int cout) { return cout >= 128 ? 128 : 64; }

static void run_conv(hipStream_t s, const bf16_t* A, const bf16_t* Bw,
                     const float* bias, float* out, const ConvDesc& d) {
  if (d.cout >= 128) {
    dim3 g((d.M + 63) / 64, d.Npad / 128);
    conv_wmma<64, 128><<<g, 256, 0, s>>>(A, Bw, bias, out, d);
  } else {
    dim3 g((d.M + 127) / 128, d.Npad / 64);
    conv_wmma<128, 64><<<g, 256, 0, s>>>(A, Bw, bias, out, d);
  }
}

static void conv3d(Ctx& c, const bf16_t* in, const float* w, int nin,
                   int cin, int cout, int k, int stride, int pad, int nout,
                   float* out) {
  int kvol = k * k * k, Kt = cin * kvol;
  int Kpad = rnd(Kt, 32), Npad = rnd(cout, blkN(cout));
  int tot = Kpad * Npad;
  pack_w3d<<<(tot + 255) / 256, 256, 0, c.s>>>(w, cout, cin, kvol, c.Wp, Kpad, Npad);
  ConvDesc d{};
  d.Din = d.Hin = d.Win = nin;
  d.Dout = d.Hout = d.Wout = nout;
  d.cin = cin; d.cout = cout;
  d.kd = d.kh = d.kw = k;
  d.stride = stride; d.pad = pad;
  d.up = 0; d.udz = d.udy = d.udx = 0;
  d.Ktot = Kt; d.Npad = Npad;
  d.M = nout * nout * nout;
  run_conv(c.s, in, c.Wp, nullptr, out, d);
}

static void bnrelu(Ctx& c, const float* t, int vox, int C, const float* g,
                   const float* b, const float* add, int relu, float* of,
                   bf16_t* ob, int obs, int obo) {
  bn_stats<<<C, 256, 0, c.s>>>(t, vox, C, c.mean, c.var);
  int tot = vox * C;
  bn_apply<<<(tot + 255) / 256, 256, 0, c.s>>>(t, c.mean, c.var, g, b, add,
                                               relu, of, ob, obs, obo, vox, C);
}

static void res_block(Ctx& c, const bf16_t* in_bf, const float* in_f32, int n,
                      int cin, int cout,
                      const float* w1, const float* g1, const float* b1,
                      const float* w2, const float* g2, const float* b2,
                      const float* ws, const float* gs, const float* bs,
                      float* out_f32, bf16_t* out_bf, int obs, int obo) {
  int vox = n * n * n;
  conv3d(c, in_bf, w1, n, cin, cout, 3, 1, 1, n, c.t0);
  bnrelu(c, c.t0, vox, cout, g1, b1, nullptr, 1, nullptr, c.hbf, cout, 0);
  conv3d(c, c.hbf, w2, n, cout, cout, 3, 1, 1, n, c.t0);
  const float* sc;
  if (ws) {
    conv3d(c, in_bf, ws, n, cin, cout, 1, 1, 0, n, c.t2);
    bnrelu(c, c.t2, vox, cout, gs, bs, nullptr, 0, c.t2, nullptr, 0, 0);
    sc = c.t2;
  } else {
    sc = in_f32;
  }
  bnrelu(c, c.t0, vox, cout, g2, b2, sc, 1, out_f32, out_bf, obs, obo);
}

static void upconv(Ctx& c, const bf16_t* in_bf, int nin, int cin, int cout,
                   const float* w, const float* g, const float* b,
                   bf16_t* out_bf, int obs) {
  int Kpad = rnd(cin, 32), Npad = rnd(cout, blkN(cout));
  int tot = 8 * Kpad * Npad;
  pack_wt<<<(tot + 255) / 256, 256, 0, c.s>>>(w, cin, cout, c.Wp, Kpad, Npad);
  int nout = 2 * nin, voxo = nout * nout * nout;
  for (int dz = 0; dz < 2; ++dz)
    for (int dy = 0; dy < 2; ++dy)
      for (int dx = 0; dx < 2; ++dx) {
        int di = dz * 4 + dy * 2 + dx;
        ConvDesc d{};
        d.Din = d.Hin = d.Win = nin;
        d.Dout = d.Hout = d.Wout = nout;
        d.cin = cin; d.cout = cout;
        d.kd = d.kh = d.kw = 1;
        d.stride = 1; d.pad = 0;
        d.up = 1; d.udz = dz; d.udy = dy; d.udx = dx;
        d.Ktot = cin; d.Npad = Npad;
        d.M = nin * nin * nin;
        run_conv(c.s, in_bf, c.Wp + (size_t)di * Kpad * Npad, nullptr, c.t0, d);
      }
  bnrelu(c, c.t0, voxo, cout, g, b, nullptr, 1, nullptr, out_bf, obs, 0);
}

extern "C" void kernel_launch(void* const* d_in, const int* in_sizes, int n_in,
                              void* d_out, int out_size, void* d_ws, size_t ws_size,
                              hipStream_t stream) {
  (void)in_sizes; (void)n_in; (void)out_size; (void)ws_size;
  const float* x = (const float*)d_in[0];
  auto P = [&](int i) { return (const float*)d_in[i]; };

  // Param indices follow JAX pytree flattening (sorted dict keys, depth-first):
  // cls(b,w)=1,2; stage1..4(down b,g,w; res1; res2); stem1; stem2; up1..4(res1;res2;t).

  const int V0 = 48 * 48 * 48, V1 = 24 * 24 * 24, V2 = 12 * 12 * 12, V3 = 6 * 6 * 6;

  char* pp = (char*)d_ws;
  auto alloc = [&](size_t bytes) -> void* {
    void* r = (void*)pp;
    pp += (bytes + 255) & ~(size_t)255;
    return r;
  };

  Ctx c; c.s = stream;
  c.t0   = (float*)alloc((size_t)V0 * 64 * 4);
  c.t2   = (float*)alloc((size_t)V0 * 64 * 4);
  c.hbf  = (bf16_t*)alloc((size_t)V0 * 64 * 2);
  c.Wp   = (bf16_t*)alloc((size_t)6 * 1024 * 1024);
  c.mean = (float*)alloc(512 * 4);
  c.var  = (float*)alloc(512 * 4);
  float*  F0  = (float*)alloc((size_t)V0 * 64 * 4);
  float*  F1  = (float*)alloc((size_t)V0 * 64 * 4);
  bf16_t* B0  = (bf16_t*)alloc((size_t)V0 * 64 * 2);
  bf16_t* B1  = (bf16_t*)alloc((size_t)V0 * 64 * 2);
  bf16_t* cc  = (bf16_t*)alloc((size_t)V0 * 128 * 2);
  bf16_t* s0  = (bf16_t*)alloc((size_t)V0 * 64 * 2);
  bf16_t* s1  = (bf16_t*)alloc((size_t)V1 * 64 * 2);
  bf16_t* s2  = (bf16_t*)alloc((size_t)V2 * 64 * 2);
  bf16_t* s3  = (bf16_t*)alloc((size_t)V3 * 128 * 2);
  bf16_t* xin = (bf16_t*)alloc((size_t)V0 * 4 * 2);

  // ---- stem ----
  to_cl_bf16<<<(V0 * 4 + 255) / 256, 256, 0, stream>>>(x, V0, 4, xin);
  conv3d(c, xin, P(71), 48, 4, 64, 3, 1, 1, 48, c.t0);
  bnrelu(c, c.t0, V0, 64, P(70), P(69), nullptr, 1, nullptr, B0, 64, 0);
  conv3d(c, B0, P(74), 48, 64, 64, 3, 1, 1, 48, c.t0);
  bnrelu(c, c.t0, V0, 64, P(73), P(72), nullptr, 1, nullptr, s0, 64, 0);

  // ---- stage1 (48->24, 64->64) ----
  conv3d(c, s0, P(5), 48, 64, 64, 2, 2, 0, 24, c.t0);
  bnrelu(c, c.t0, V1, 64, P(4), P(3), nullptr, 1, F0, B0, 64, 0);
  res_block(c, B0, F0, 24, 64, 64, P(10), P(8), P(6), P(11), P(9), P(7),
            nullptr, nullptr, nullptr, F1, B1, 64, 0);
  res_block(c, B1, F1, 24, 64, 64, P(16), P(14), P(12), P(17), P(15), P(13),
            nullptr, nullptr, nullptr, nullptr, s1, 64, 0);

  // ---- stage2 (24->12, 64->64) ----
  conv3d(c, s1, P(20), 24, 64, 64, 2, 2, 0, 12, c.t0);
  bnrelu(c, c.t0, V2, 64, P(19), P(18), nullptr, 1, F0, B0, 64, 0);
  res_block(c, B0, F0, 12, 64, 64, P(25), P(23), P(21), P(26), P(24), P(22),
            nullptr, nullptr, nullptr, F1, B1, 64, 0);
  res_block(c, B1, F1, 12, 64, 64, P(31), P(29), P(27), P(32), P(30), P(28),
            nullptr, nullptr, nullptr, nullptr, s2, 64, 0);

  // ---- stage3 (12->6, 64->128) ----
  conv3d(c, s2, P(35), 12, 64, 64, 2, 2, 0, 6, c.t0);
  bnrelu(c, c.t0, V3, 64, P(34), P(33), nullptr, 1, F0, B0, 64, 0);
  res_block(c, B0, F0, 6, 64, 128, P(42), P(39), P(36), P(43), P(40), P(37),
            P(44), P(41), P(38), F1, B1, 128, 0);
  res_block(c, B1, F1, 6, 128, 128, P(49), P(47), P(45), P(50), P(48), P(46),
            nullptr, nullptr, nullptr, nullptr, s3, 128, 0);

  // ---- stage4 (6->3, 128->256) ----
  conv3d(c, s3, P(53), 6, 128, 128, 2, 2, 0, 3, c.t0);
  bnrelu(c, c.t0, 27, 128, P(52), P(51), nullptr, 1, F0, B0, 128, 0);
  res_block(c, B0, F0, 3, 128, 256, P(60), P(57), P(54), P(61), P(58), P(55),
            P(62), P(59), P(56), F1, B1, 256, 0);
  res_block(c, B1, F1, 3, 256, 256, P(67), P(65), P(63), P(68), P(66), P(64),
            nullptr, nullptr, nullptr, nullptr, B0, 256, 0);

  // ---- up1 (3->6, 256 + skip128 -> 256) ----
  upconv(c, B0, 3, 256, 256, P(92), P(91), P(90), cc, 384);
  copy_ch<<<(V3 * 128 + 255) / 256, 256, 0, stream>>>(s3, 128, cc, 384, 256, V3);
  res_block(c, cc, nullptr, 6, 384, 256, P(81), P(78), P(75), P(82), P(79), P(76),
            P(83), P(80), P(77), F0, B0, 256, 0);
  res_block(c, B0, F0, 6, 256, 256, P(88), P(86), P(84), P(89), P(87), P(85),
            nullptr, nullptr, nullptr, nullptr, B1, 256, 0);

  // ---- up2 (6->12, 128 + skip64 -> 128) ----
  upconv(c, B1, 6, 256, 128, P(110), P(109), P(108), cc, 192);
  copy_ch<<<(V2 * 64 + 255) / 256, 256, 0, stream>>>(s2, 64, cc, 192, 128, V2);
  res_block(c, cc, nullptr, 12, 192, 128, P(99), P(96), P(93), P(100), P(97), P(94),
            P(101), P(98), P(95), F0, B0, 128, 0);
  res_block(c, B0, F0, 12, 128, 128, P(106), P(104), P(102), P(107), P(105), P(103),
            nullptr, nullptr, nullptr, nullptr, B1, 128, 0);

  // ---- up3 (12->24, 64 + skip64 -> 64) ----
  upconv(c, B1, 12, 128, 64, P(128), P(127), P(126), cc, 128);
  copy_ch<<<(V1 * 64 + 255) / 256, 256, 0, stream>>>(s1, 64, cc, 128, 64, V1);
  res_block(c, cc, nullptr, 24, 128, 64, P(117), P(114), P(111), P(118), P(115), P(112),
            P(119), P(116), P(113), F0, B0, 64, 0);
  res_block(c, B0, F0, 24, 64, 64, P(124), P(122), P(120), P(125), P(123), P(121),
            nullptr, nullptr, nullptr, nullptr, B1, 64, 0);

  // ---- up4 (24->48, 64 + skip64 -> 64) ----
  upconv(c, B1, 24, 64, 64, P(146), P(145), P(144), cc, 128);
  copy_ch<<<(V0 * 64 + 255) / 256, 256, 0, stream>>>(s0, 64, cc, 128, 64, V0);
  res_block(c, cc, nullptr, 48, 128, 64, P(135), P(132), P(129), P(136), P(133), P(130),
            P(137), P(134), P(131), F0, B0, 64, 0);
  res_block(c, B0, F0, 48, 64, 64, P(142), P(140), P(138), P(143), P(141), P(139),
            nullptr, nullptr, nullptr, nullptr, B1, 64, 0);

  // ---- classifier: (V0 x 64) @ (64 x 19) + b -> d_out ----
  {
    int Kpad = 64, Npad = 64;
    pack_rm<<<(Kpad * Npad + 255) / 256, 256, 0, stream>>>(P(2), 64, 19, c.Wp, Kpad, Npad);
    ConvDesc d{};
    d.Din = d.Hin = d.Win = 48;
    d.Dout = d.Hout = d.Wout = 48;
    d.cin = 64; d.cout = 19;
    d.kd = d.kh = d.kw = 1;
    d.stride = 1; d.pad = 0;
    d.up = 0; d.udz = d.udy = d.udx = 0;
    d.Ktot = 64; d.Npad = Npad;
    d.M = V0;
    run_conv(stream, B1, c.Wp, P(1), (float*)d_out, d);
  }
}
